// SelfAttention_5875515261294
// MI455X (gfx1250) — matrix operand order repaired and checked by reading, not compiled, not run on hardware
//
#include <hip/hip_runtime.h>
#include <hip/hip_bf16.h>

// ---------- types ----------
typedef __bf16 v16bf __attribute__((ext_vector_type(16)));
typedef __bf16 v8bf  __attribute__((ext_vector_type(8)));
typedef __bf16 v4bf  __attribute__((ext_vector_type(4)));
typedef float  v8f   __attribute__((ext_vector_type(8)));

#define D_MODEL 1024
#define N_HEADS 16
#define D_K     64
#define T_SEQ   2048
#define B_SZ    2
#define M_ROWS  (B_SZ * T_SEQ)   // 4096

static __device__ __forceinline__ v8f vzero8() {
    v8f z = {0.f, 0.f, 0.f, 0.f, 0.f, 0.f, 0.f, 0.f};
    return z;
}

static __device__ __forceinline__ __bf16 f2bf(float f) {
    unsigned u = __builtin_bit_cast(unsigned, f);
    unsigned r = (u + 0x7FFFu + ((u >> 16) & 1u)) >> 16;
    unsigned short s = (unsigned short)r;
    return __builtin_bit_cast(__bf16, s);
}

static __device__ __forceinline__ v16bf load_frag(const __bf16* p0, const __bf16* p1) {
    union { struct { v8bf lo; v8bf hi; } s; v16bf v; } u;
    u.s.lo = *reinterpret_cast<const v8bf*>(p0);
    u.s.hi = *reinterpret_cast<const v8bf*>(p1);
    return u.v;
}

static __device__ __forceinline__ v8f wmma_bf16(v16bf a, v16bf b, v8f c) {
    return __builtin_amdgcn_wmma_f32_16x16x32_bf16(false, a, false, b, (short)0, c, false, false);
}

static __device__ __forceinline__ float rowmax16(float v) {
    v = fmaxf(v, __shfl_xor(v, 1, 32));
    v = fmaxf(v, __shfl_xor(v, 2, 32));
    v = fmaxf(v, __shfl_xor(v, 4, 32));
    v = fmaxf(v, __shfl_xor(v, 8, 32));
    return v;
}
static __device__ __forceinline__ float rowsum16(float v) {
    v += __shfl_xor(v, 1, 32);
    v += __shfl_xor(v, 2, 32);
    v += __shfl_xor(v, 4, 32);
    v += __shfl_xor(v, 8, 32);
    return v;
}

// ---------- kernel 1: fp32 -> bf16 conversion (one-time) ----------
__global__ void sa_cvt_bf16(const float* __restrict__ in, __bf16* __restrict__ out, int n4) {
    int i = blockIdx.x * blockDim.x + threadIdx.x;
    if (i < n4) {
        const float4 f = reinterpret_cast<const float4*>(in)[i];
        v4bf o;
        o.x = f2bf(f.x); o.y = f2bf(f.y); o.z = f2bf(f.z); o.w = f2bf(f.w);
        reinterpret_cast<v4bf*>(out)[i] = o;
    }
}

// ---------- kernel 2: fused QKV projection GEMM (bf16 WMMA) ----------
// C[m,e] = sum_k xb[m,k] * W[e,k] + b[e]  (torch Linear: x @ W^T + b)
// Each wave: 32M x 64N tile (2 A frags, 4 B frags, 8 accumulators per K-step).
// blockIdx.z selects {Q,K,V}. Q,K stored [B,H,T,dk]; V stored transposed [B,H,dk,T].
__global__ __launch_bounds__(256) void sa_qkv_gemm(
    const __bf16* __restrict__ xb,
    const __bf16* __restrict__ wq, const __bf16* __restrict__ wk, const __bf16* __restrict__ wv,
    const float* __restrict__ bq, const float* __restrict__ bk, const float* __restrict__ bv,
    __bf16* __restrict__ Qb, __bf16* __restrict__ Kb, __bf16* __restrict__ Vt) {

    const int z    = blockIdx.z;
    const int lane = threadIdx.x & 31;
    const int wv_  = threadIdx.x >> 5;     // wave id 0..7
    const int ln   = lane & 15;
    const int lh   = lane >> 4;

    const int m0 = blockIdx.y * 256 + wv_ * 32;
    const int n0 = blockIdx.x * 64;

    const __bf16* wsel = (z == 0) ? wq : (z == 1) ? wk : wv;
    const float*  bsel = (z == 0) ? bq : (z == 1) ? bk : bv;

    v8f acc[2][4];
    #pragma unroll
    for (int i = 0; i < 2; ++i)
        #pragma unroll
        for (int j = 0; j < 4; ++j) acc[i][j] = vzero8();

    const __bf16* arow0 = xb + (size_t)(m0 + ln) * D_MODEL;
    const __bf16* arow1 = arow0 + (size_t)16 * D_MODEL;

    for (int kk = 0; kk < D_MODEL; kk += 32) {
        // gather all fragments for this K-step first (batched loads, MLP)
        const v16bf a0 = load_frag(arow0 + kk + lh * 8, arow0 + kk + 16 + lh * 8);
        const v16bf a1 = load_frag(arow1 + kk + lh * 8, arow1 + kk + 16 + lh * 8);
        v16bf bf[4];
        #pragma unroll
        for (int j = 0; j < 4; ++j) {
            const __bf16* brow = wsel + (size_t)(n0 + j * 16 + ln) * D_MODEL + kk + lh * 8;
            bf[j] = load_frag(brow, brow + 16);
        }
        #pragma unroll
        for (int j = 0; j < 4; ++j) {
            acc[0][j] = wmma_bf16(a0, bf[j], acc[0][j]);
            acc[1][j] = wmma_bf16(a1, bf[j], acc[1][j]);
        }
    }

    #pragma unroll
    for (int j = 0; j < 4; ++j) {
        const int e    = n0 + j * 16 + ln;
        const float bi = bsel[e];
        const int h    = e >> 6;      // e / 64
        const int d    = e & 63;
        #pragma unroll
        for (int i = 0; i < 2; ++i) {
            #pragma unroll
            for (int r = 0; r < 8; ++r) {
                const int m  = m0 + i * 16 + r + 8 * lh;
                const int bb = m >> 11;   // m / 2048
                const int t  = m & 2047;
                const __bf16 val = f2bf(acc[i][j][r] + bi);
                if (z == 0) {
                    Qb[(((size_t)(bb * N_HEADS + h) * T_SEQ) + t) * D_K + d] = val;
                } else if (z == 1) {
                    Kb[(((size_t)(bb * N_HEADS + h) * T_SEQ) + t) * D_K + d] = val;
                } else {
                    Vt[(((size_t)(bb * N_HEADS + h) * D_K) + d) * T_SEQ + t] = val;
                }
            }
        }
    }
}

// ---------- kernel 3: causal flash attention (bf16 WMMA, online softmax) ----------
// One wave = 16 query rows of one (b,h). 4 waves per block.
__global__ __launch_bounds__(128) void sa_attn(
    const __bf16* __restrict__ Qb, const __bf16* __restrict__ Kb,
    const __bf16* __restrict__ Vt, __bf16* __restrict__ Ob) {

    const int lane = threadIdx.x & 31;
    const int w    = threadIdx.x >> 5;     // 0..3
    const int ln   = lane & 15;
    const int lh   = lane >> 4;
    const int bh   = blockIdx.y;           // b*16 + h
    const int b    = bh >> 4;
    const int h    = bh & 15;
    const int t0   = blockIdx.x * 64 + w * 16;

    __shared__ __bf16 pbuf[4][16 * 40];    // per-wave 16x32 P tile, stride 40 (16B aligned rows)

    // Q fragments, resident across the whole key loop
    const __bf16* qbase = Qb + ((size_t)bh * T_SEQ + t0 + ln) * D_K;
    const v16bf qf0 = load_frag(qbase + lh * 8,      qbase + 16 + lh * 8);   // dk  0..31
    const v16bf qf1 = load_frag(qbase + 32 + lh * 8, qbase + 48 + lh * 8);   // dk 32..63

    v8f o[4];
    #pragma unroll
    for (int j = 0; j < 4; ++j) o[j] = vzero8();

    float mrow[8], lrow[8], alpha[8];
    #pragma unroll
    for (int r = 0; r < 8; ++r) { mrow[r] = -1e30f; lrow[r] = 0.f; }

    const float scale = 0.125f;            // 1/sqrt(64)

    for (int s0 = 0; s0 < t0 + 16; s0 += 32) {
        // ---- S = Q K^T, two 16-key score tiles ----
        const __bf16* kb0 = Kb + ((size_t)bh * T_SEQ + s0 + ln) * D_K + lh * 8;
        const __bf16* kb1 = kb0 + (size_t)16 * D_K;
        const v16bf k00 = load_frag(kb0,      kb0 + 16);
        const v16bf k01 = load_frag(kb0 + 32, kb0 + 48);
        const v16bf k10 = load_frag(kb1,      kb1 + 16);
        const v16bf k11 = load_frag(kb1 + 32, kb1 + 48);
        v8f s0a = vzero8(), s1a = vzero8();
        s0a = wmma_bf16(qf0, k00, s0a);
        s0a = wmma_bf16(qf1, k01, s0a);
        s1a = wmma_bf16(qf0, k10, s1a);
        s1a = wmma_bf16(qf1, k11, s1a);

        // ---- causal mask + online softmax (rows: VGPR r -> t0 + r + 8*lh) ----
        #pragma unroll
        for (int r = 0; r < 8; ++r) {
            const int trow = t0 + r + 8 * lh;
            float x0 = s0a[r] * scale;
            float x1 = s1a[r] * scale;
            if (s0 + ln      > trow) x0 = -1e30f;
            if (s0 + 16 + ln > trow) x1 = -1e30f;
            const float tmax = rowmax16(fmaxf(x0, x1));
            const float mnew = fmaxf(mrow[r], tmax);
            alpha[r] = __expf(mrow[r] - mnew);
            const float p0 = __expf(x0 - mnew);
            const float p1 = __expf(x1 - mnew);
            lrow[r] = lrow[r] * alpha[r] + rowsum16(p0 + p1);
            mrow[r] = mnew;
            const int row = r + 8 * lh;
            pbuf[w][row * 40 + ln]      = f2bf(p0);
            pbuf[w][row * 40 + 16 + ln] = f2bf(p1);
        }

        // rescale running output
        #pragma unroll
        for (int j = 0; j < 4; ++j)
            #pragma unroll
            for (int r = 0; r < 8; ++r) o[j][r] *= alpha[r];

        // in-wave LDS store->load turnaround (HW keeps LDS in order; fence the compiler too)
        asm volatile("s_wait_dscnt 0" ::: "memory");

        // ---- P fragment from LDS (A layout: row ln, K chunks at lh*8 / 16+lh*8) ----
        const __bf16* pb = &pbuf[w][ln * 40 + lh * 8];
        const v16bf pf = load_frag(pb, pb + 16);

        // ---- O += P V  (B frags from V^T: row d contiguous along keys) ----
        const __bf16* vb = Vt + ((size_t)bh * D_K + ln) * T_SEQ + s0 + lh * 8;
        v16bf vf[4];
        #pragma unroll
        for (int j = 0; j < 4; ++j) {
            const __bf16* vj = vb + (size_t)(j * 16) * T_SEQ;
            vf[j] = load_frag(vj, vj + 16);
        }
        #pragma unroll
        for (int j = 0; j < 4; ++j)
            o[j] = wmma_bf16(pf, vf[j], o[j]);
    }

    // ---- epilogue: O /= l, write bf16 [B,T,D] ----
    #pragma unroll
    for (int r = 0; r < 8; ++r) {
        const float inv = 1.0f / lrow[r];
        const int trow  = t0 + r + 8 * lh;
        #pragma unroll
        for (int j = 0; j < 4; ++j) {
            const int e = h * D_K + j * 16 + ln;
            Ob[((size_t)(b * T_SEQ + trow)) * D_MODEL + e] = f2bf(o[j][r] * inv);
        }
    }
}

// ---------- kernel 4: output projection GEMM (bf16 WMMA -> fp32 out) ----------
__global__ __launch_bounds__(256) void sa_out_gemm(
    const __bf16* __restrict__ Ob, const __bf16* __restrict__ wo,
    const float* __restrict__ bo, float* __restrict__ out) {

    const int lane = threadIdx.x & 31;
    const int wv_  = threadIdx.x >> 5;
    const int ln   = lane & 15;
    const int lh   = lane >> 4;
    const int m0   = blockIdx.y * 256 + wv_ * 32;
    const int n0   = blockIdx.x * 64;

    v8f acc[2][4];
    #pragma unroll
    for (int i = 0; i < 2; ++i)
        #pragma unroll
        for (int j = 0; j < 4; ++j) acc[i][j] = vzero8();

    const __bf16* arow0 = Ob + (size_t)(m0 + ln) * D_MODEL;
    const __bf16* arow1 = arow0 + (size_t)16 * D_MODEL;

    for (int kk = 0; kk < D_MODEL; kk += 32) {
        const v16bf a0 = load_frag(arow0 + kk + lh * 8, arow0 + kk + 16 + lh * 8);
        const v16bf a1 = load_frag(arow1 + kk + lh * 8, arow1 + kk + 16 + lh * 8);
        v16bf bf[4];
        #pragma unroll
        for (int j = 0; j < 4; ++j) {
            const __bf16* brow = wo + (size_t)(n0 + j * 16 + ln) * D_MODEL + kk + lh * 8;
            bf[j] = load_frag(brow, brow + 16);
        }
        #pragma unroll
        for (int j = 0; j < 4; ++j) {
            acc[0][j] = wmma_bf16(a0, bf[j], acc[0][j]);
            acc[1][j] = wmma_bf16(a1, bf[j], acc[1][j]);
        }
    }

    #pragma unroll
    for (int j = 0; j < 4; ++j) {
        const int e    = n0 + j * 16 + ln;
        const float bi = bo[e];
        #pragma unroll
        for (int i = 0; i < 2; ++i) {
            #pragma unroll
            for (int r = 0; r < 8; ++r) {
                const int m = m0 + i * 16 + r + 8 * lh;
                out[(size_t)m * D_MODEL + e] = acc[i][j][r] + bi;
            }
        }
    }
}

// ---------- launch ----------
extern "C" void kernel_launch(void* const* d_in, const int* in_sizes, int n_in,
                              void* d_out, int out_size, void* d_ws, size_t ws_size,
                              hipStream_t stream) {
    const float* x  = (const float*)d_in[0];
    const float* WQ = (const float*)d_in[1];
    const float* bQ = (const float*)d_in[2];
    const float* WK = (const float*)d_in[3];
    const float* bK = (const float*)d_in[4];
    const float* WV = (const float*)d_in[5];
    const float* bV = (const float*)d_in[6];
    const float* WO = (const float*)d_in[7];
    const float* bO = (const float*)d_in[8];
    float* out = (float*)d_out;

    const size_t X_ELEMS = (size_t)M_ROWS * D_MODEL;        // 4,194,304
    const size_t W_ELEMS = (size_t)D_MODEL * D_MODEL;       // 1,048,576
    const size_t QKV_ELEMS = (size_t)B_SZ * N_HEADS * T_SEQ * D_K; // 4,194,304

    char* ws = (char*)d_ws;
    size_t off = 0;
    __bf16* xb  = (__bf16*)(ws + off); off += X_ELEMS * 2;
    __bf16* wqb = (__bf16*)(ws + off); off += W_ELEMS * 2;
    __bf16* wkb = (__bf16*)(ws + off); off += W_ELEMS * 2;
    __bf16* wvb = (__bf16*)(ws + off); off += W_ELEMS * 2;
    __bf16* wob = (__bf16*)(ws + off); off += W_ELEMS * 2;
    __bf16* Qb  = (__bf16*)(ws + off); off += QKV_ELEMS * 2;
    __bf16* Kb  = (__bf16*)(ws + off); off += QKV_ELEMS * 2;
    __bf16* Vt  = (__bf16*)(ws + off); off += QKV_ELEMS * 2;
    __bf16* Ob  = (__bf16*)(ws + off); off += X_ELEMS * 2;

    // 1) one-time fp32->bf16 conversions
    sa_cvt_bf16<<<dim3((unsigned)(X_ELEMS / 4 / 256)), dim3(256), 0, stream>>>(x,  xb,  (int)(X_ELEMS / 4));
    sa_cvt_bf16<<<dim3((unsigned)(W_ELEMS / 4 / 256)), dim3(256), 0, stream>>>(WQ, wqb, (int)(W_ELEMS / 4));
    sa_cvt_bf16<<<dim3((unsigned)(W_ELEMS / 4 / 256)), dim3(256), 0, stream>>>(WK, wkb, (int)(W_ELEMS / 4));
    sa_cvt_bf16<<<dim3((unsigned)(W_ELEMS / 4 / 256)), dim3(256), 0, stream>>>(WV, wvb, (int)(W_ELEMS / 4));
    sa_cvt_bf16<<<dim3((unsigned)(W_ELEMS / 4 / 256)), dim3(256), 0, stream>>>(WO, wob, (int)(W_ELEMS / 4));

    // 2) QKV projections: grid (N/64, M/256, 3)
    sa_qkv_gemm<<<dim3(D_MODEL / 64, M_ROWS / 256, 3), dim3(256), 0, stream>>>(
        xb, wqb, wkb, wvb, bQ, bK, bV, Qb, Kb, Vt);

    // 3) causal flash attention: grid (T/64, B*H), 4 waves/block
    sa_attn<<<dim3(T_SEQ / 64, B_SZ * N_HEADS), dim3(128), 0, stream>>>(Qb, Kb, Vt, Ob);

    // 4) output projection -> fp32 d_out
    sa_out_gemm<<<dim3(D_MODEL / 64, M_ROWS / 256), dim3(256), 0, stream>>>(Ob, wob, bO, out);
}
